// Attention_88974542504050
// MI455X (gfx1250) — compile-verified, matrix operand order checked by reading, not run on hardware
//
#include <hip/hip_runtime.h>
#include <hip/hip_bf16.h>
#include <cstdint>

#define EN_HID 1024
#define DE_HID 1024
#define BATCH 64
#define SRC_LEN 2048

typedef __attribute__((ext_vector_type(16))) __bf16 v16bf;
typedef __attribute__((ext_vector_type(8)))  float  v8f;
typedef __attribute__((ext_vector_type(4)))  __bf16 bf4;

union FragU { v16bf v; uint4 u[2]; };

__device__ __forceinline__ float tanh_fast(float x) {
  // tanh via single v_exp_f32; clamp avoids inf/inf
  float xc = fminf(fmaxf(x, -8.0f), 8.0f);
  float e  = __expf(2.0f * xc);
  return (e - 1.0f) / (e + 1.0f);
}

// ---------------------------------------------------------------------------
// Kernel 1: pre[b][d] = sum_h hidden[b,h] * w_de_w[d,h] + w_de_b[d] + w_en_b[d]
// ---------------------------------------------------------------------------
__global__ void __launch_bounds__(256)
pre_kernel(const float* __restrict__ hidden, const float* __restrict__ wde,
           const float* __restrict__ bde, const float* __restrict__ ben,
           float* __restrict__ pre)
{
  __shared__ float hs[DE_HID];
  const int b = blockIdx.x, tid = threadIdx.x;
  for (int i = tid; i < DE_HID / 4; i += 256)
    *(float4*)(hs + i * 4) = *(const float4*)(hidden + (size_t)b * DE_HID + i * 4);
  __syncthreads();
  for (int j = 0; j < 4; ++j) {
    const int d = tid + j * 256;
    const float* row = wde + (size_t)d * DE_HID;
    float acc = 0.0f;
    for (int h = 0; h < DE_HID; h += 4) {
      float4 w4 = *(const float4*)(row + h);
      acc += w4.x * hs[h] + w4.y * hs[h + 1] + w4.z * hs[h + 2] + w4.w * hs[h + 3];
    }
    pre[(size_t)b * DE_HID + d] = acc + bde[d] + ben[d];
  }
}

// ---------------------------------------------------------------------------
// Kernel 2: convert w_en_w (fp32, [Dd][E]) -> bf16 (stays resident in L2)
// ---------------------------------------------------------------------------
__global__ void __launch_bounds__(256)
convw_kernel(const float* __restrict__ w, __bf16* __restrict__ wbf)
{
  const size_t i4 = (size_t)blockIdx.x * 256 + threadIdx.x;
  float4 f = *(const float4*)(w + i4 * 4);
  bf4 h;
  h.x = (__bf16)f.x; h.y = (__bf16)f.y; h.z = (__bf16)f.z; h.w = (__bf16)f.w;
  *(bf4*)(wbf + i4 * 4) = h;
}

// ---------------------------------------------------------------------------
// Kernel 3: WMMA GEMM + tanh + v-dot -> scores[b][s]
// WG = (b, 128-row s-tile). A tile (128x1024 bf16 = 258KB) staged in LDS.
// 8 waves; wave w owns N columns [chunk*128 + w*16, +16); K loop 32/step.
// All 8 A fragments loaded into distinct regs before the 8 wmma so the
// matrix ops issue back-to-back behind a single s_wait_dscnt.
// ---------------------------------------------------------------------------
__global__ void __launch_bounds__(256)
scores_kernel(const float* __restrict__ en, const __bf16* __restrict__ wbf,
              const float* __restrict__ pre, const float* __restrict__ vw,
              float* __restrict__ scores)
{
  extern __shared__ char smem[];
  __bf16* As = (__bf16*)smem;                       // [128][1032] (pad 8 halves)
  float*  Sc = (float*)(smem + 128 * 1032 * 2);     // [128] score accumulators

  const int tid = threadIdx.x;
  const int b   = blockIdx.y;
  const int s0  = blockIdx.x * 128;

  // Stage A tile: fp32 global -> bf16 LDS (16B ds stores)
  const float* enB = en + ((size_t)b * SRC_LEN + s0) * EN_HID;
  for (int idx = tid; idx < 128 * 128; idx += 256) {
    const int row = idx >> 7;
    const int c8  = (idx & 127) << 3;
    float4 f0 = *(const float4*)(enB + (size_t)row * EN_HID + c8);
    float4 f1 = *(const float4*)(enB + (size_t)row * EN_HID + c8 + 4);
    union { bf4 h[2]; uint4 u; } p;
    p.h[0].x = (__bf16)f0.x; p.h[0].y = (__bf16)f0.y;
    p.h[0].z = (__bf16)f0.z; p.h[0].w = (__bf16)f0.w;
    p.h[1].x = (__bf16)f1.x; p.h[1].y = (__bf16)f1.y;
    p.h[1].z = (__bf16)f1.z; p.h[1].w = (__bf16)f1.w;
    *(uint4*)(As + row * 1032 + c8) = p.u;
  }
  if (tid < 128) Sc[tid] = 0.0f;
  __syncthreads();

  const int wave = tid >> 5, lane = tid & 31;
  const int hl = lane >> 4, l16 = lane & 15;

  for (int chunk = 0; chunk < 8; ++chunk) {
    const int n   = chunk * 128 + wave * 16 + l16;   // this lane's N column
    const float vn  = vw[n];
    const float prn = pre[(size_t)b * DE_HID + n];

    v8f c[8];
    #pragma unroll
    for (int st = 0; st < 8; ++st)
      #pragma unroll
      for (int i = 0; i < 8; ++i) c[st][i] = 0.0f;

    const __bf16* wrow = wbf + (size_t)n * EN_HID;
    for (int k0 = 0; k0 < EN_HID; k0 += 32) {
      // B fragment: lane = N column, K split across lane halves (mirrors A layout)
      FragU bfrag;
      bfrag.u[0] = *(const uint4*)(wrow + k0 + hl * 8);
      bfrag.u[1] = *(const uint4*)(wrow + k0 + hl * 8 + 16);

      // Load ALL A fragments first (independent regs -> one dscnt wait)
      FragU af[8];
      #pragma unroll
      for (int st = 0; st < 8; ++st) {
        const __bf16* arow = As + (st * 16 + l16) * 1032 + k0 + hl * 8;
        af[st].u[0] = *(const uint4*)(arow);
        af[st].u[1] = *(const uint4*)(arow + 16);
      }
      // Back-to-back wmma; each chain only depends on its own accumulator
      #pragma unroll
      for (int st = 0; st < 8; ++st) {
        c[st] = __builtin_amdgcn_wmma_f32_16x16x32_bf16(
            false, af[st].v, false, bfrag.v, (short)0, c[st], false, false);
      }
    }

    // scores[m] += sum_n vn * tanh(C[m][n] + pre[n]); N striped over 16 lanes
    #pragma unroll
    for (int st = 0; st < 8; ++st) {
      #pragma unroll
      for (int r = 0; r < 8; ++r) {
        float val = vn * tanh_fast(c[st][r] + prn);
        val += __shfl_xor(val, 1, 16);
        val += __shfl_xor(val, 2, 16);
        val += __shfl_xor(val, 4, 16);
        val += __shfl_xor(val, 8, 16);
        if (l16 == 0)  // lane 0 -> M=r, lane 16 -> M=r+8 (C layout)
          atomicAdd(&Sc[st * 16 + r + 8 * hl], val);
      }
    }
  }
  __syncthreads();
  if (tid < 128) scores[(size_t)b * SRC_LEN + s0 + tid] = Sc[tid];
}

// ---------------------------------------------------------------------------
// Kernel 4: softmax over S per batch
// ---------------------------------------------------------------------------
__global__ void __launch_bounds__(256)
softmax_kernel(const float* __restrict__ scores, float* __restrict__ attn)
{
  __shared__ float red[256];
  const int b = blockIdx.x, tid = threadIdx.x;
  const float* s = scores + (size_t)b * SRC_LEN;
  float* a = attn + (size_t)b * SRC_LEN;

  float m = -3.4e38f;
  for (int i = tid; i < SRC_LEN; i += 256) m = fmaxf(m, s[i]);
  red[tid] = m; __syncthreads();
  for (int o = 128; o > 0; o >>= 1) {
    if (tid < o) red[tid] = fmaxf(red[tid], red[tid + o]);
    __syncthreads();
  }
  const float M = red[0];
  __syncthreads();

  float sum = 0.0f;
  for (int i = tid; i < SRC_LEN; i += 256) {
    float e = __expf(s[i] - M);
    a[i] = e;
    sum += e;
  }
  red[tid] = sum; __syncthreads();
  for (int o = 128; o > 0; o >>= 1) {
    if (tid < o) red[tid] += red[tid + o];
    __syncthreads();
  }
  const float inv = 1.0f / red[0];
  for (int i = tid; i < SRC_LEN; i += 256) a[i] *= inv;
}

// ---------------------------------------------------------------------------
// Kernel 5: partial weighted sums over 128-row S chunks (streams en_output)
// ---------------------------------------------------------------------------
__global__ void __launch_bounds__(256)
wsum_partial_kernel(const float* __restrict__ en, const float* __restrict__ attn,
                    float* __restrict__ partial)
{
  const int b = blockIdx.x, ch = blockIdx.y, tid = threadIdx.x;
  const int e = tid * 4;
  float4 acc = {0.f, 0.f, 0.f, 0.f};
  const float* aB  = attn + (size_t)b * SRC_LEN + ch * 128;
  const float* enB = en + ((size_t)b * SRC_LEN + (size_t)ch * 128) * EN_HID;
  for (int s = 0; s < 128; ++s) {
    const float a = aB[s];
    float4 v = *(const float4*)(enB + (size_t)s * EN_HID + e);
    acc.x += a * v.x; acc.y += a * v.y; acc.z += a * v.z; acc.w += a * v.w;
  }
  *(float4*)(partial + ((size_t)(b * 16 + ch)) * EN_HID + e) = acc;
}

// ---------------------------------------------------------------------------
// Kernel 6: reduce partials -> weighted[b][e]
// ---------------------------------------------------------------------------
__global__ void __launch_bounds__(256)
wsum_final_kernel(const float* __restrict__ partial, float* __restrict__ out)
{
  const int b = blockIdx.x, tid = threadIdx.x;
  const int e = tid * 4;
  float4 acc = {0.f, 0.f, 0.f, 0.f};
  for (int c = 0; c < 16; ++c) {
    float4 v = *(const float4*)(partial + ((size_t)(b * 16 + c)) * EN_HID + e);
    acc.x += v.x; acc.y += v.y; acc.z += v.z; acc.w += v.w;
  }
  *(float4*)(out + (size_t)b * EN_HID + e) = acc;
}

// ---------------------------------------------------------------------------
extern "C" void kernel_launch(void* const* d_in, const int* in_sizes, int n_in,
                              void* d_out, int out_size, void* d_ws, size_t ws_size,
                              hipStream_t stream) {
  (void)in_sizes; (void)n_in; (void)out_size; (void)ws_size;
  const float* hidden = (const float*)d_in[0];
  const float* en     = (const float*)d_in[1];
  const float* w_en_w = (const float*)d_in[2];
  const float* w_en_b = (const float*)d_in[3];
  const float* w_de_w = (const float*)d_in[4];
  const float* w_de_b = (const float*)d_in[5];
  const float* v_w    = (const float*)d_in[6];

  float* out = (float*)d_out;
  float* weighted = out;                       // (B,1,E) = 64*1024
  float* attn     = out + BATCH * EN_HID;      // (B,S)   = 64*2048

  char* ws = (char*)d_ws;                      // ~6.8 MB total
  float*  pre     = (float*)(ws);                                        // 256 KB
  __bf16* wbf     = (__bf16*)(ws + (256 << 10));                         // 2 MB
  float*  scores  = (float*)(ws + (256 << 10) + (2 << 20));              // 512 KB
  float*  partial = (float*)(ws + (256 << 10) + (2 << 20) + (512 << 10)); // 4 MB

  pre_kernel<<<dim3(BATCH), 256, 0, stream>>>(hidden, w_de_w, w_de_b, w_en_b, pre);
  convw_kernel<<<dim3((DE_HID * EN_HID) / 1024), 256, 0, stream>>>(w_en_w, wbf);

  const size_t smem = (size_t)128 * 1032 * 2 + 128 * 4;  // 258 KB (< 320 KB WGP LDS)
  scores_kernel<<<dim3(SRC_LEN / 128, BATCH), 256, smem, stream>>>(en, wbf, pre, v_w, scores);

  softmax_kernel<<<dim3(BATCH), 256, 0, stream>>>(scores, attn);
  wsum_partial_kernel<<<dim3(BATCH, 16), 256, 0, stream>>>(en, attn, partial);
  wsum_final_kernel<<<dim3(BATCH), 256, 0, stream>>>(partial, weighted);
}
